// GINNet_41120016892603
// MI455X (gfx1250) — matrix operand order repaired and checked by reading, not compiled, not run on hardware
//
#include <hip/hip_runtime.h>
#include <hip/hip_bf16.h>

// ---------------------------------------------------------------------------
// GIN (5x GINConv + BN) + global_add_pool + MLP head for MI455X (gfx1250).
// z = h + sum_{j->i} h_j materialized via init-copy + float atomics (the
// memory-bound part: ~2.2 GB traffic -> ~95us @ 23.3 TB/s HBM); MLPs run on
// v_wmma_f32_16x16x32_f16 with branch-free b128 fragment loads.
// ---------------------------------------------------------------------------

typedef __attribute__((ext_vector_type(16))) _Float16 v16h;
typedef __attribute__((ext_vector_type(8)))  _Float16 v8h;
typedef __attribute__((ext_vector_type(8)))  float    v8f;

#define F_IN 37
#define DIM  32
#define BN_EPS 1e-5f

// ---------------- z init: z[row, 0..LDZ) = h[row, 0..F) padded with 0 -------
template<int LDZ>
__global__ __launch_bounds__(256) void gin_init_z_kernel(
    const float* __restrict__ h, float* __restrict__ z, int N, int F, int ldh)
{
    int tid = blockIdx.x * blockDim.x + threadIdx.x;
    int row = tid / LDZ;
    int c   = tid & (LDZ - 1);
    if (row >= N) return;
    z[(size_t)row * LDZ + c] = (c < F) ? h[(size_t)row * ldh + c] : 0.0f;
}

// ---------------- edge scatter-add: z[dst] += h[src] ------------------------
__global__ __launch_bounds__(256) void gin_scatter_kernel(
    const float* __restrict__ h, const int* __restrict__ src,
    const int* __restrict__ dst, float* __restrict__ z,
    long total, int F, int ldh, int ldz)
{
    long tid = (long)blockIdx.x * blockDim.x + threadIdx.x;
    if (tid >= total) return;
    int e = (int)(tid / F);
    int f = (int)(tid - (long)e * F);
    atomicAdd(&z[(long)dst[e] * ldz + f], h[(long)src[e] * ldh + f]);
}

// ---------------- fused GIN MLP + BN layer (WMMA) ---------------------------
// z is [N x KP] with cols >= K zero.  h_out = BN(relu(relu(z@Wa+ba)@Wb+bb)).
// One wave32 per 16-row tile; two 16x16 WMMA accumulators cover the 32 cols.
template<int K, int KP>
__global__ __launch_bounds__(128) void gin_mlp_bn_kernel(
    const float* __restrict__ z,
    const float* __restrict__ Wa,  const float* __restrict__ ba,
    const float* __restrict__ Wb,  const float* __restrict__ bb,
    const float* __restrict__ bng, const float* __restrict__ bnb,
    const float* __restrict__ bnm, const float* __restrict__ bnv,
    float* __restrict__ hout, int nRows)
{
    constexpr int WAVES = 4;
    // transposed weight staging: sWaT[n*KP + k] = Wa[k][n]  (B frags contiguous)
    __shared__ alignas(32) _Float16 sWaT[DIM * KP];
    __shared__ alignas(32) _Float16 sWbT[DIM * DIM];
    __shared__ alignas(32) _Float16 sY1[WAVES][16 * DIM];

    const int tid = threadIdx.x;
    #pragma unroll 4
    for (int i = tid; i < DIM * KP; i += 128) {
        int n = i / KP, k = i - n * KP;
        sWaT[i] = (k < K) ? (_Float16)Wa[k * DIM + n] : (_Float16)0.0f;
    }
    #pragma unroll 4
    for (int i = tid; i < DIM * DIM; i += 128) {
        int n = i >> 5, k = i & 31;
        sWbT[i] = (_Float16)Wb[k * DIM + n];
    }
    __syncthreads();

    const int wave = tid >> 5;
    const int lane = tid & 31;
    const int r0   = (blockIdx.x * WAVES + wave) * 16;
    if (r0 >= nRows) return;

    const int M   = lane & 15;        // A-frag row / D-frag column
    const int hi  = lane >> 4;
    const int kbA = hi * 16;          // A fragment K base
    const int kbB = hi * 16;          // B fragment K base
    const int row = min(r0 + M, nRows - 1);   // clamp; stores are guarded
    const float* zrow = z + (size_t)row * KP;

    // ---- matmul 1: z[16xKP] @ Wa[KPx32] ----
    v8f c0 = {}; v8f c1 = {};
    #pragma unroll
    for (int kc = 0; kc < KP; kc += 32) {
        float4 p0 = *(const float4*)(zrow + kc + kbA);
        float4 p1 = *(const float4*)(zrow + kc + kbA + 4);
        float4 p2 = *(const float4*)(zrow + kc + kbA + 8);
        float4 p3 = *(const float4*)(zrow + kc + kbA + 12);
        v16h a;
        a[0]=(_Float16)p0.x;  a[1]=(_Float16)p0.y;  a[2]=(_Float16)p0.z;  a[3]=(_Float16)p0.w;
        a[4]=(_Float16)p1.x;  a[5]=(_Float16)p1.y;  a[6]=(_Float16)p1.z;  a[7]=(_Float16)p1.w;
        a[8]=(_Float16)p2.x;  a[9]=(_Float16)p2.y;  a[10]=(_Float16)p2.z; a[11]=(_Float16)p2.w;
        a[12]=(_Float16)p3.x; a[13]=(_Float16)p3.y; a[14]=(_Float16)p3.z; a[15]=(_Float16)p3.w;
        v16h b0 = *(const v16h*)(sWaT + (size_t)M        * KP + kc + kbB);
        v16h b1 = *(const v16h*)(sWaT + (size_t)(16 + M) * KP + kc + kbB);
        c0 = __builtin_amdgcn_wmma_f32_16x16x32_f16(false, a, false, b0,
                                                    (short)0, c0, false, false);
        c1 = __builtin_amdgcn_wmma_f32_16x16x32_f16(false, a, false, b1,
                                                    (short)0, c1, false, false);
    }

    // ---- bias + relu, stage hidden tile to LDS (re-layout D -> A frag) ----
    _Float16* y1 = sY1[wave];
    const int rOfs = hi * 8;          // D frag: VGPR r -> row r + 8*hi
    {
        float bia0 = ba[M], bia1 = ba[16 + M];
        #pragma unroll
        for (int r = 0; r < 8; ++r) {
            float v0 = c0[r] + bia0; v0 = v0 > 0.0f ? v0 : 0.0f;
            float v1 = c1[r] + bia1; v1 = v1 > 0.0f ? v1 : 0.0f;
            y1[(rOfs + r) * DIM + M]      = (_Float16)v0;
            y1[(rOfs + r) * DIM + 16 + M] = (_Float16)v1;
        }
    }
    // LDS ops are in-order per wave; block compiler reordering + wait returns.
    asm volatile("s_wait_dscnt 0x0" ::: "memory");

    // ---- matmul 2: y1[16x32] @ Wb[32x32] ----
    v8h alo = *(const v8h*)(y1 + M * DIM + kbA);
    v8h ahi = *(const v8h*)(y1 + M * DIM + kbA + 8);
    v16h a2;
    #pragma unroll
    for (int e = 0; e < 8; ++e) { a2[e] = alo[e]; a2[8 + e] = ahi[e]; }
    v16h b20 = *(const v16h*)(sWbT + (size_t)M        * DIM + kbB);
    v16h b21 = *(const v16h*)(sWbT + (size_t)(16 + M) * DIM + kbB);
    v8f d0 = {}; v8f d1 = {};
    d0 = __builtin_amdgcn_wmma_f32_16x16x32_f16(false, a2, false, b20,
                                                (short)0, d0, false, false);
    d1 = __builtin_amdgcn_wmma_f32_16x16x32_f16(false, a2, false, b21,
                                                (short)0, d1, false, false);

    // ---- bias + relu + BatchNorm(eval) -> hout ----
    float s0 = bng[M]      * rsqrtf(bnv[M]      + BN_EPS);
    float s1 = bng[16 + M] * rsqrtf(bnv[16 + M] + BN_EPS);
    float t0 = bnb[M]      - bnm[M]      * s0;
    float t1 = bnb[16 + M] - bnm[16 + M] * s1;
    float bb0 = bb[M], bb1 = bb[16 + M];
    #pragma unroll
    for (int r = 0; r < 8; ++r) {
        int orow = r0 + rOfs + r;
        if (orow >= nRows) continue;
        float v0 = d0[r] + bb0; v0 = v0 > 0.0f ? v0 : 0.0f;
        float v1 = d1[r] + bb1; v1 = v1 > 0.0f ? v1 : 0.0f;
        hout[(size_t)orow * DIM + M]      = v0 * s0 + t0;
        hout[(size_t)orow * DIM + 16 + M] = v1 * s1 + t1;
    }
}

// ---------------- global add pool -------------------------------------------
__global__ __launch_bounds__(256) void gin_pool_kernel(
    const float* __restrict__ h, const int* __restrict__ batch,
    float* __restrict__ pooled, int N)
{
    int tid  = blockIdx.x * blockDim.x + threadIdx.x;
    int node = tid >> 5, f = tid & 31;
    if (node >= N) return;
    atomicAdd(&pooled[(long)batch[node] * DIM + f], h[(size_t)node * DIM + f]);
}

__global__ __launch_bounds__(256) void gin_zero_kernel(float* __restrict__ p, int n)
{
    int i = blockIdx.x * blockDim.x + threadIdx.x;
    if (i < n) p[i] = 0.0f;
}

// ---------------- classifier head: fc1 -> relu -> fc2 -> log_softmax --------
__global__ __launch_bounds__(64) void gin_head_kernel(
    const float* __restrict__ pooled,
    const float* __restrict__ fc1W, const float* __restrict__ fc1b,
    const float* __restrict__ fc2W, const float* __restrict__ fc2b,
    float* __restrict__ out, int G)
{
    int g = blockIdx.x * blockDim.x + threadIdx.x;
    if (g >= G) return;
    const float* p = pooled + (size_t)g * DIM;
    float z[DIM];
    #pragma unroll 4
    for (int j = 0; j < DIM; ++j) {
        float acc = fc1b[j];
        for (int k = 0; k < DIM; ++k) acc += p[k] * fc1W[k * DIM + j];
        z[j] = acc > 0.0f ? acc : 0.0f;
    }
    float l0 = fc2b[0], l1 = fc2b[1];
    for (int k = 0; k < DIM; ++k) { l0 += z[k] * fc2W[k * 2 + 0];
                                    l1 += z[k] * fc2W[k * 2 + 1]; }
    float m = fmaxf(l0, l1);
    float lse = m + logf(expf(l0 - m) + expf(l1 - m));
    out[(size_t)g * 2 + 0] = l0 - lse;
    out[(size_t)g * 2 + 1] = l1 - lse;
}

// ---------------------------------------------------------------------------
extern "C" void kernel_launch(void* const* d_in, const int* in_sizes, int n_in,
                              void* d_out, int out_size, void* d_ws, size_t ws_size,
                              hipStream_t stream)
{
    const int N = in_sizes[0] / F_IN;     // 100000
    const int E = in_sizes[1] / 2;        // 1600000
    const int G = out_size / 2;           // 2000

    const float* x     = (const float*)d_in[0];
    const int*   ei    = (const int*)d_in[1];
    const int*   src   = ei;
    const int*   dst   = ei + E;
    const int*   batch = (const int*)d_in[2];

    // params: d_in[3 + i*8 + {Wa,ba,Wb,bb,bn_g,bn_b,bn_m,bn_v}], fc at 43..46
    auto LP = [&](int layer, int j) { return (const float*)d_in[3 + layer * 8 + j]; };
    const float* fc1W = (const float*)d_in[43];
    const float* fc1b = (const float*)d_in[44];
    const float* fc2W = (const float*)d_in[45];
    const float* fc2b = (const float*)d_in[46];

    // workspace layout (floats): zbuf[N*64] | hA[N*32] | hB[N*32] | pooled[G*32]
    float* zbuf   = (float*)d_ws;
    float* hA     = zbuf + (size_t)N * 64;
    float* hB     = hA   + (size_t)N * DIM;
    float* pooled = hB   + (size_t)N * DIM;

    const int tiles     = (N + 15) / 16;
    const int mlpBlocks = (tiles + 3) / 4;

    // ---- layer 0 (K = 37, padded to 64) ----
    {
        int initThreads = N * 64;
        gin_init_z_kernel<64><<<(initThreads + 255) / 256, 256, 0, stream>>>(
            x, zbuf, N, F_IN, F_IN);
        long total = (long)E * F_IN;
        gin_scatter_kernel<<<(int)((total + 255) / 256), 256, 0, stream>>>(
            x, src, dst, zbuf, total, F_IN, F_IN, 64);
        gin_mlp_bn_kernel<F_IN, 64><<<mlpBlocks, 128, 0, stream>>>(
            zbuf, LP(0,0), LP(0,1), LP(0,2), LP(0,3),
            LP(0,4), LP(0,5), LP(0,6), LP(0,7), hA, N);
    }

    // ---- layers 1..4 (K = KP = 32) ----
    float* hin = hA; float* hout = hB;
    for (int i = 1; i < 5; ++i) {
        int initThreads = N * DIM;
        gin_init_z_kernel<DIM><<<(initThreads + 255) / 256, 256, 0, stream>>>(
            hin, zbuf, N, DIM, DIM);
        long total = (long)E * DIM;
        gin_scatter_kernel<<<(int)((total + 255) / 256), 256, 0, stream>>>(
            hin, src, dst, zbuf, total, DIM, DIM, DIM);
        gin_mlp_bn_kernel<DIM, DIM><<<mlpBlocks, 128, 0, stream>>>(
            zbuf, LP(i,0), LP(i,1), LP(i,2), LP(i,3),
            LP(i,4), LP(i,5), LP(i,6), LP(i,7), hout, N);
        float* t = hin; hin = hout; hout = t;
    }

    // ---- pool + head ----
    gin_zero_kernel<<<(G * DIM + 255) / 256, 256, 0, stream>>>(pooled, G * DIM);
    gin_pool_kernel<<<(N * DIM + 255) / 256, 256, 0, stream>>>(hin, batch, pooled, N);
    gin_head_kernel<<<(G + 63) / 64, 64, 0, stream>>>(
        pooled, fc1W, fc1b, fc2W, fc2b, (float*)d_out, G);
}
